// LocBlock2dNT_75196287419028
// MI455X (gfx1250) — matrix operand order repaired and checked by reading, not compiled, not run on hardware
//
#include <hip/hip_runtime.h>

typedef _Float16 h16 __attribute__((ext_vector_type(16)));
typedef _Float16 h8  __attribute__((ext_vector_type(8)));
typedef _Float16 h4  __attribute__((ext_vector_type(4)));
typedef float    f4  __attribute__((ext_vector_type(4)));
typedef float    v8f __attribute__((ext_vector_type(8)));

// LDS row stride in halves: 64 data + 8 pad = 144 bytes -> the 16 lanes of a
// half-wave hit 16 distinct 4-DWORD bank groups on ds_load_b128 (stride 36
// DWORDs, gcd(36,64)=4 => conflict-free).
#define LSTR 72

// Problem constants (from reference): N=64, C=64, D=64, P=Q=16, f=4, O=256, K=1024
// x strides (floats): n:262144 c:4096 row:64
// w strides (floats): o:262144 c:4096 (p*16+q):16 f:1
// out index: (n*256 + o)*256 + pq

__device__ __forceinline__ void stage_store(_Float16* __restrict__ A,
                                            _Float16* __restrict__ B,
                                            int nb, int aSt, int bSt,
                                            f4 a0, f4 a1, f4 a2, f4 a3,
                                            f4 b0, f4 b1, f4 b2, f4 b3)
{
    *(h4*)&A[(nb +  0) * LSTR + aSt] = __builtin_convertvector(a0, h4);
    *(h4*)&A[(nb + 16) * LSTR + aSt] = __builtin_convertvector(a1, h4);
    *(h4*)&A[(nb + 32) * LSTR + aSt] = __builtin_convertvector(a2, h4);
    *(h4*)&A[(nb + 48) * LSTR + aSt] = __builtin_convertvector(a3, h4);
    h4 c0 = __builtin_convertvector(b0, h4);
    h4 c1 = __builtin_convertvector(b1, h4);
    h4 c2 = __builtin_convertvector(b2, h4);
    h4 c3 = __builtin_convertvector(b3, h4);
    *(h8*)&B[bSt]     = __builtin_shufflevector(c0, c1, 0,1,2,3,4,5,6,7);
    *(h8*)&B[bSt + 8] = __builtin_shufflevector(c2, c3, 0,1,2,3,4,5,6,7);
}

__device__ __forceinline__ h16 cat16(h8 lo, h8 hi)
{
    return __builtin_shufflevector(lo, hi, 0,1,2,3,4,5,6,7,8,9,10,11,12,13,14,15);
}

__global__ __launch_bounds__(256) void locblock_f16_wmma(
    const float* __restrict__ x,   // (64,64,64,64)
    const float* __restrict__ w,   // (256,64,16,16,16)
    float* __restrict__ out)       // (64,256,16,16)
{
    __shared__ __align__(16) _Float16 lsA[2][64 * LSTR];
    __shared__ __align__(16) _Float16 lsB[2][64 * LSTR];

    const int tid = threadIdx.x;
    const int pq  = blockIdx.x;            // p*16 + q
    const int p   = pq >> 4;
    const int o0  = blockIdx.y << 6;       // output-channel tile base

    // ---- global -> register staging maps (one K=64 chunk = 4 channels) ----
    // A side: thread covers (n = nb+16j, cc = ccA, fr), loads 4 contiguous floats (fc)
    const int fr  = tid & 3;
    const int ccA = (tid >> 2) & 3;
    const int nb  = tid >> 4;
    const float* aPtr = x + ((long)nb * 262144 + (long)ccA * 4096
                             + (long)((p << 2) + fr) * 64 + (long)((pq & 15) << 2));
    // B side: thread covers (o = ob, cc = ccB), loads 16 contiguous floats (f)
    const int ob  = tid >> 2;
    const int ccB = tid & 3;
    const float* bPtr = w + ((long)(o0 + ob) * 262144 + (long)ccB * 4096
                             + (long)pq * 16);

    const int aSt = ccA * 16 + fr * 4;     // + row*LSTR  (k = cc*16 + fr*4 + fc)
    const int bSt = ob * LSTR + ccB * 16;  // k = cc*16 + f

    // ---- wave fragment maps (A/B: contiguous k = 16*half + s) ----
    const int lane = tid & 31;
    const int wm   = (tid >> 5) & 3;       // M tile (16 rows)
    const int wn   = tid >> 7;             // N slab (32 cols)
    const int half = lane >> 4;
    const int l    = lane & 15;
    const int aFr  = (wm * 16 + l) * LSTR + half * 16;  // A: row M=l, {K0..15|K16..31} by half
    const int bFr  = (wn * 32 + l) * LSTR + half * 16;  // B: row N=l, {K0..15|K16..31} by half

    v8f acc0 = {}, acc1 = {};
    f4 aR0, aR1, aR2, aR3, bR0, bR1, bR2, bR3;

    // ---- prologue: chunk 0 global loads + LDS stage ----
    aR0 = *(const f4*)(aPtr);
    aR1 = *(const f4*)(aPtr + 16l * 262144);
    aR2 = *(const f4*)(aPtr + 32l * 262144);
    aR3 = *(const f4*)(aPtr + 48l * 262144);
    bR0 = *(const f4*)(bPtr + 0);
    bR1 = *(const f4*)(bPtr + 4);
    bR2 = *(const f4*)(bPtr + 8);
    bR3 = *(const f4*)(bPtr + 12);
    stage_store(lsA[0], lsB[0], nb, aSt, bSt, aR0, aR1, aR2, aR3, bR0, bR1, bR2, bR3);
    __syncthreads();

    int buf = 0;
    for (int ch = 0; ch < 16; ++ch) {
        const bool more = ch < 15;
        if (more) {                               // prefetch chunk ch+1 (c0 advances by 4 -> 16384 floats)
            const float* ap = aPtr + (long)(ch + 1) * 16384;
            aR0 = *(const f4*)(ap);
            aR1 = *(const f4*)(ap + 16l * 262144);
            aR2 = *(const f4*)(ap + 32l * 262144);
            aR3 = *(const f4*)(ap + 48l * 262144);
            const float* bp = bPtr + (long)(ch + 1) * 16384;
            bR0 = *(const f4*)(bp + 0);
            bR1 = *(const f4*)(bp + 4);
            bR2 = *(const f4*)(bp + 8);
            bR3 = *(const f4*)(bp + 12);
        }
#pragma unroll
        for (int ks = 0; ks < 2; ++ks) {          // two K=32 steps per chunk
            const int ko = ks << 5;
            const _Float16* Ab = &lsA[buf][aFr + ko];
            h16 a = cat16(*(const h8*)(Ab), *(const h8*)(Ab + 8));
            const _Float16* Bb = &lsB[buf][bFr + ko];
            h16 b = cat16(*(const h8*)(Bb), *(const h8*)(Bb + 8));
            acc0 = __builtin_amdgcn_wmma_f32_16x16x32_f16(
                       false, a, false, b, (short)0, acc0, false, false);
            const _Float16* Bc = Bb + 16 * LSTR;
            h16 b2 = cat16(*(const h8*)(Bc), *(const h8*)(Bc + 8));
            acc1 = __builtin_amdgcn_wmma_f32_16x16x32_f16(
                       false, a, false, b2, (short)0, acc1, false, false);
        }
        __syncthreads();                          // all waves done reading buf^1's producer barrier
        if (more)
            stage_store(lsA[buf ^ 1], lsB[buf ^ 1], nb, aSt, bSt,
                        aR0, aR1, aR2, aR3, bR0, bR1, bR2, bR3);
        buf ^= 1;
        __syncthreads();                          // stores to new buf visible
    }

    // ---- epilogue: relu(acc/32); C/D layout: M = r + 8*half, N = l ----
    const float scale = 1.0f / 32.0f;
    const int ocol = o0 + wn * 32 + l;
#pragma unroll
    for (int r = 0; r < 8; ++r) {
        const int n = wm * 16 + half * 8 + r;
        float v0 = acc0[r] * scale;  v0 = v0 > 0.0f ? v0 : 0.0f;
        float v1 = acc1[r] * scale;  v1 = v1 > 0.0f ? v1 : 0.0f;
        out[(long)(n * 256 + ocol) * 256 + pq]        = v0;
        out[(long)(n * 256 + ocol + 16) * 256 + pq]   = v1;
    }
}

extern "C" void kernel_launch(void* const* d_in, const int* in_sizes, int n_in,
                              void* d_out, int out_size, void* d_ws, size_t ws_size,
                              hipStream_t stream) {
    const float* x = (const float*)d_in[0];   // (64,64,64,64) fp32
    const float* w = (const float*)d_in[1];   // (256,64,16,16,16) fp32
    float* out = (float*)d_out;               // (64,256,16,16) fp32
    dim3 grid(256, 4);                        // 256 spatial sites x 4 o-tiles
    locblock_f16_wmma<<<grid, dim3(256), 0, stream>>>(x, w, out);
}
